// PW_Quad_31310311588558
// MI455X (gfx1250) — matrix operand order repaired and checked by reading, not compiled, not run on hardware
//
#include <hip/hip_runtime.h>
#include <hip/hip_bf16.h>

typedef __attribute__((ext_vector_type(16))) _Float16 v16h;
typedef __attribute__((ext_vector_type(8)))  float    v8f;

#define BROWS   65536
#define TT      8
#define KBINS   64
#define HID     256
#define OUTC    1032
#define OUTP    1040      // padded to 65 tiles of 16
#define MROWS   32        // rows per block
#define KSTEPS  8         // 256 / 32

union AFr { v16h v; unsigned u[8]; _Float16 h[16]; };
union CFr { v8f  v; float f[8]; };

// ---------------- prep: f16 convert + WMMA-B-fragment swizzle ----------------
// B fragment (32x16 f16, ld via two b128/lane): lane l -> n = ntile*16+(l&15),
// halfs j=0..15 -> K = kstep*32 + (l>>4)*16 + j. Stored fragment-linear.
__global__ void prep_kernel(const float* __restrict__ W2,
                            const float* __restrict__ W3,
                            const float* __restrict__ b3,
                            _Float16* __restrict__ W2f,
                            _Float16* __restrict__ W3f,
                            float* __restrict__ b3p) {
  int idx = blockIdx.x * blockDim.x + threadIdx.x;
  if (idx < HID * HID) {                       // W2: 16 ntiles * 8 ksteps
    int j = idx & 15, lane = (idx >> 4) & 31, f = idx >> 9;
    int ntile = f >> 3, kstep = f & 7;
    int n = ntile * 16 + (lane & 15);
    int k = kstep * 32 + (lane >> 4) * 16 + j;
    W2f[idx] = (_Float16)W2[k * HID + n];
  }
  if (idx < HID * OUTP) {                      // W3: 65 ntiles * 8 ksteps (padded)
    int j = idx & 15, lane = (idx >> 4) & 31, f = idx >> 9;
    int ntile = f >> 3, kstep = f & 7;
    int n = ntile * 16 + (lane & 15);
    int k = kstep * 32 + (lane >> 4) * 16 + j;
    W3f[idx] = (n < OUTC) ? (_Float16)W3[k * OUTC + n] : (_Float16)0.0f;
  }
  if (idx < OUTP) b3p[idx] = (idx < OUTC) ? b3[idx] : 0.0f;
}

// A fragment (16x32 f16) from LDS row-major [m][256]:
// lanes 0-15: V0..7 = K 0..15 ; lanes 16-31: +16
__device__ inline v16h load_afrag(const _Float16* __restrict__ hbuf,
                                  int lane, int mtile, int kstep) {
  AFr a;
  int m  = mtile * 16 + (lane & 15);
  int kb = ((lane >> 4) << 4) + kstep * 32;
  const unsigned* row = (const unsigned*)(hbuf + m * HID);
#pragma unroll
  for (int v = 0; v < 8; ++v) {
    int k = (2 * v) + kb;   // even -> dword aligned
    a.u[v] = row[k >> 1];
  }
  return a.v;
}

// B fragment from prepacked global: 32 contiguous bytes per lane
__device__ inline v16h load_bfrag(const _Float16* __restrict__ Wf,
                                  int lane, int ntile, int kstep) {
  union { v16h v; uint4 q[2]; } b;
  const uint4* p = (const uint4*)(Wf + (((size_t)(ntile * KSTEPS + kstep) * 32 + lane) << 4));
  b.q[0] = p[0];
  b.q[1] = p[1];
  return b.v;
}

// ---------------- fused: h1 -> GEMM2(relu) -> GEMM3 -> spline ----------------
__global__ __launch_bounds__(256)
void fused_kernel(const float* __restrict__ x,  const float* __restrict__ W1,
                  const float* __restrict__ b1, const float* __restrict__ b2,
                  const _Float16* __restrict__ W2f, const _Float16* __restrict__ W3f,
                  const float* __restrict__ b3p, float* __restrict__ out) {
  extern __shared__ char smem[];
  _Float16* h1     = (_Float16*)(smem);                       // 32*256*2 = 16384
  _Float16* h2     = (_Float16*)(smem + 16384);               // 16384
  float*    params = (float*)   (smem + 32768);               // 32*1040*4 = 133120
  float*    xls    = (float*)   (smem + 165888);              // 32*16*4   = 2048
  float*    logq   = (float*)   (smem + 167936);              // 32*8*4    = 1024

  const int tid  = threadIdx.x;
  const int lane = tid & 31;
  const int wave = tid >> 5;
  const int r0   = blockIdx.x * MROWS;

  // ---- stage x tile ----
  for (int i = tid; i < MROWS * 16; i += 256) xls[i] = x[(size_t)r0 * 16 + i];
  __syncthreads();

  // ---- h1 = xc @ W1 + b1 (K=8, scalar; W1 is 8KB, L2/L0 hot) ----
  for (int o = tid; o < MROWS * HID; o += 256) {
    int m = o >> 8, n = o & 255;
    float acc = b1[n];
#pragma unroll
    for (int j = 0; j < TT; ++j) acc += xls[m * 16 + 2 * j] * W1[j * HID + n];
    h1[m * HID + n] = (_Float16)acc;
  }
  __syncthreads();

  // ---- GEMM2: h2 = relu(h1 @ W2 + b2), 2 N-tiles per wave ----
  for (int nt = wave * 2; nt < wave * 2 + 2; ++nt) {
    v8f acc0 = {}, acc1 = {};
#pragma unroll
    for (int ks = 0; ks < KSTEPS; ++ks) {
      v16h bf = load_bfrag(W2f, lane, nt, ks);
      v16h a0 = load_afrag(h1, lane, 0, ks);
      v16h a1 = load_afrag(h1, lane, 1, ks);
      acc0 = __builtin_amdgcn_wmma_f32_16x16x32_f16(false, a0, false, bf, (short)0, acc0, false, false);
      acc1 = __builtin_amdgcn_wmma_f32_16x16x32_f16(false, a1, false, bf, (short)0, acc1, false, false);
    }
    int n = nt * 16 + (lane & 15);
    float bias = b2[n];
    int mb = (lane >> 4) << 3;
    CFr c0, c1; c0.v = acc0; c1.v = acc1;
#pragma unroll
    for (int v = 0; v < 8; ++v) {
      float v0 = c0.f[v] + bias; v0 = v0 > 0.0f ? v0 : 0.0f;
      float v1 = c1.f[v] + bias; v1 = v1 > 0.0f ? v1 : 0.0f;
      h2[(mb + v)      * HID + n] = (_Float16)v0;
      h2[(16 + mb + v) * HID + n] = (_Float16)v1;
    }
  }
  __syncthreads();

  // ---- GEMM3: params = h2 @ W3 + b3 (65 N-tiles over 8 waves) ----
  for (int nt = wave; nt < OUTP / 16; nt += 8) {
    v8f acc0 = {}, acc1 = {};
#pragma unroll
    for (int ks = 0; ks < KSTEPS; ++ks) {
      v16h bf = load_bfrag(W3f, lane, nt, ks);
      v16h a0 = load_afrag(h2, lane, 0, ks);
      v16h a1 = load_afrag(h2, lane, 1, ks);
      acc0 = __builtin_amdgcn_wmma_f32_16x16x32_f16(false, a0, false, bf, (short)0, acc0, false, false);
      acc1 = __builtin_amdgcn_wmma_f32_16x16x32_f16(false, a1, false, bf, (short)0, acc1, false, false);
    }
    int n = nt * 16 + (lane & 15);
    float bias = b3p[n];
    int mb = (lane >> 4) << 3;
    CFr c0, c1; c0.v = acc0; c1.v = acc1;
#pragma unroll
    for (int v = 0; v < 8; ++v) {
      params[(mb + v)      * OUTP + n] = c0.f[v] + bias;
      params[(16 + mb + v) * OUTP + n] = c1.f[v] + bias;
    }
  }
  __syncthreads();

  // ---- spline epilogue: one thread per (row m, transform t) ----
  {
    int m = tid >> 3, t = tid & 7;
    const float* Vun = params + m * OUTP + t * (KBINS + 1);
    const float* Wun = params + m * OUTP + TT * (KBINS + 1) + t * KBINS;

    float wmax = -3.4e38f;
    for (int k = 0; k < KBINS; ++k) wmax = fmaxf(wmax, Wun[k]);
    float wsum = 0.0f;
    for (int k = 0; k < KBINS; ++k) wsum += __expf(Wun[k] - wmax);
    float winv = 1.0f / wsum;

    float xt = xls[m * 16 + 2 * t + 1];
    float denom = 0.0f, cum = 0.0f, cint = 0.0f;
    float evp = __expf(Vun[0]);
    int found = 0;
    float cumbefore = 0.0f, wbb = 0.0f, evb = 0.0f, evb1 = 0.0f;
    for (int k = 0; k < KBINS; ++k) {
      float wb  = __expf(Wun[k] - wmax) * winv;
      float ev  = __expf(Vun[k + 1]);
      float seg = 0.5f * (evp + ev) * wb;
      denom += seg;
      float cnew = cum + wb;
      if (!found) {
        if (cnew > xt) { found = 1; cumbefore = cum; wbb = wb; evb = evp; evb1 = ev; }
        else cint += seg;
      }
      cum = cnew; evp = ev;
    }
    if (!found) {  // argmax fallback -> bin 0
      cumbefore = 0.0f; cint = 0.0f;
      wbb  = __expf(Wun[0] - wmax) * winv;
      evb  = __expf(Vun[0]);
      evb1 = __expf(Vun[1]);
    }
    float dinv  = 1.0f / denom;
    float alpha = (xt - cumbefore) / wbb;
    float Vb = evb * dinv, Vb1 = evb1 * dinv;
    float C  = 0.5f * alpha * alpha * (Vb1 - Vb) * wbb + alpha * Vb * wbb + cint * dinv;
    float q  = Vb + alpha * (Vb1 - Vb);
    logq[m * 8 + t] = __logf(q);

    size_t row = (size_t)(r0 + m);
    float zc = xls[m * 16 + 2 * t]; zc = zc > 1.0f ? 1.0f : zc;
    float zt = C > 1.0f ? 1.0f : C;
    out[row * 16 + 2 * t]     = zc;
    out[row * 16 + 2 * t + 1] = zt;
  }
  __syncthreads();

  // ---- log_J reduction: one thread per row ----
  if (tid < MROWS) {
    float s = 0.0f;
#pragma unroll
    for (int t = 0; t < TT; ++t) s += logq[tid * 8 + t];
    out[(size_t)BROWS * 16 + r0 + tid] = s;
  }
}

extern "C" void kernel_launch(void* const* d_in, const int* in_sizes, int n_in,
                              void* d_out, int out_size, void* d_ws, size_t ws_size,
                              hipStream_t stream) {
  const float* x  = (const float*)d_in[0];
  const float* W1 = (const float*)d_in[1];
  const float* b1 = (const float*)d_in[2];
  const float* W2 = (const float*)d_in[3];
  const float* b2 = (const float*)d_in[4];
  const float* W3 = (const float*)d_in[5];
  const float* b3 = (const float*)d_in[6];

  char* ws = (char*)d_ws;
  _Float16* W2f = (_Float16*)(ws);                         // 256*256*2  = 131072 B
  _Float16* W3f = (_Float16*)(ws + 131072);                // 256*1040*2 = 532480 B
  float*    b3p = (float*)   (ws + 131072 + 532480);       // 1040*4     = 4160 B

  prep_kernel<<<(HID * OUTP + 255) / 256, 256, 0, stream>>>(W2, W3, b3, W2f, W3f, b3p);

  size_t smem = 16384 + 16384 + 133120 + 2048 + 1024;      // 168960 B (< 320KB/WGP)
  fused_kernel<<<BROWS / MROWS, 256, smem, stream>>>(x, W1, b1, b2, W2f, W3f, b3p,
                                                     (float*)d_out);
}
